// VersorLinear_62113817034792
// MI455X (gfx1250) — compile-verified, matrix operand order checked by reading, not run on hardware
//
#include <hip/hip_runtime.h>
#include <hip/hip_fp16.h>

// MI455X / gfx1250 (CDNA5, wave32) Cl(4,1) geometric linear layer.
// y = normalize( X[4096x4096] @ B[4096x4096] ),
// B[f*32+i, o*32+k] = W[o,f,i^k] * cayley_sign(i, i^k).
//
// Fast path (needs 64MB workspace): precompute xh = f16(x) and Bt = B^T in f16
// once, then run a double-buffered WMMA GEMM whose tiles are staged with
// gfx1250 async global->LDS copies (ASYNCcnt), normalization fused in the
// epilogue.  Fallback path: round-1 fused kernel (no workspace).

typedef __attribute__((ext_vector_type(16))) _Float16 v16h;
typedef __attribute__((ext_vector_type(8)))  float    v8f;

#define IN_F   128
#define OUT_F  128
#define GA_DIM 32
#define KDIM   (IN_F * GA_DIM)   // 4096 reduction length
#define NDIM   (OUT_F * GA_DIM)  // 4096 output columns

#define TILE_M 128
#define TILE_N 128
#define TILE_K 64
#define NKTILE (KDIM / TILE_K)   // 64
#define LDH    72                // padded halves per LDS row (16B-aligned rows)

union FragU { uint4 q[2]; v16h v; };
union PackU { __half2 h2[4]; uint4 u; };

// Cayley sign parity for e_a * e_b in Cl(4,1): reordering swaps + metric (e5^2=-1).
__device__ __forceinline__ int ga_parity(int a, int b) {
  return (__popc((a >> 1) & b) + __popc((a >> 2) & b)
        + __popc((a >> 3) & b) + __popc((a >> 4) & b)
        + ((a & b) >> 4)) & 1;
}

__device__ __forceinline__ unsigned sign_mask(int k) {
  unsigned sgn = 0u;
  #pragma unroll
  for (int i = 0; i < GA_DIM; ++i)
    sgn |= (unsigned)ga_parity(i, i ^ k) << i;
  return sgn;
}

__device__ __forceinline__ float sflip(float w, unsigned sgn, int i) {
  return __uint_as_float(__float_as_uint(w) ^ (((sgn >> i) & 1u) << 31));
}

// one async 16B global->LDS copy; OFF applies to both global and LDS address
#define ASYNC_B128(lds_off, gptr, OFF)                                        \
  asm volatile("global_load_async_to_lds_b128 %0, %1, off offset:" #OFF       \
               :: "v"(lds_off), "v"(gptr) : "memory")

// ---------------------------------------------------------------------------
// Phase 1a: x f32 -> f16
__global__ __launch_bounds__(256)
void cvt_x_kernel(const float* __restrict__ in, __half* __restrict__ out, int n8) {
  int idx = blockIdx.x * blockDim.x + threadIdx.x;
  if (idx >= n8) return;
  const float4* src = (const float4*)in + (size_t)idx * 2;
  float4 f0 = src[0], f1 = src[1];
  PackU p;
  p.h2[0] = __floats2half2_rn(f0.x, f0.y);
  p.h2[1] = __floats2half2_rn(f0.z, f0.w);
  p.h2[2] = __floats2half2_rn(f1.x, f1.y);
  p.h2[3] = __floats2half2_rn(f1.z, f1.w);
  ((uint4*)out)[idx] = p.u;
}

// Phase 1b: Bt[c, r] = sign(i, i^k) * W[o, f, i^k]  (c = o*32+k, r = f*32+i), f16
__global__ __launch_bounds__(256)
void expand_w_kernel(const float* __restrict__ wg, __half* __restrict__ bt) {
  const int c = blockIdx.x;            // 0..4095
  const int o = c >> 5, k = c & 31;
  const unsigned sgn = sign_mask(k);
  const int t = threadIdx.x;           // 0..255, 16 elements each
  const int f = t >> 1;
  const int ibase = (t & 1) << 4;
  const float* wrow = wg + (size_t)o * KDIM + f * GA_DIM;
  __half* dst = bt + (size_t)c * KDIM + t * 16;
  #pragma unroll
  for (int e = 0; e < 16; e += 2) {
    int i0 = ibase + e;
    float w0 = sflip(wrow[(i0    ) ^ k], sgn, i0    );
    float w1 = sflip(wrow[(i0 + 1) ^ k], sgn, i0 + 1);
    *(__half2*)(dst + e) = __floats2half2_rn(w0, w1);
  }
}

// ---------------------------------------------------------------------------
// Phase 2: double-buffered async-copy WMMA GEMM + fused normalization
__global__ __launch_bounds__(128)
void versor_gemm_async(const __half* __restrict__ xh,  // [B,4096] f16 row-major
                       const __half* __restrict__ bt,  // [4096,4096] f16 (B^T)
                       float* __restrict__ yg)
{
  __shared__ __half Ash[2][TILE_M * LDH];
  __shared__ __half Bsh[2][TILE_N * LDH];

  const int t    = threadIdx.x;     // 0..127
  const int lane = t & 31;
  const int wv   = t >> 5;
  const int wm   = (wv >> 1) << 6;
  const int wn   = (wv &  1) << 6;
  const int h    = lane >> 4;
  const int l16  = lane & 15;

  const int bm0 = blockIdx.y * TILE_M;
  const int bn0 = blockIdx.x * TILE_N;

  // per-thread copy jobs: row t of the A tile and row t of the Bt tile (128B each)
  const __half* asrc = xh + (size_t)(bm0 + t) * KDIM;
  const __half* bsrc = bt + (size_t)(bn0 + t) * KDIM;
  const unsigned aoff[2] = { (unsigned)(uintptr_t)&Ash[0][t * LDH],
                             (unsigned)(uintptr_t)&Ash[1][t * LDH] };
  const unsigned boff[2] = { (unsigned)(uintptr_t)&Bsh[0][t * LDH],
                             (unsigned)(uintptr_t)&Bsh[1][t * LDH] };

  auto issue_tile = [&](int buf, int k0) {
    const __half* ga = asrc + k0;
    const __half* gb = bsrc + k0;
    unsigned la = aoff[buf], lb = boff[buf];
    ASYNC_B128(la, ga,   0); ASYNC_B128(la, ga,  16);
    ASYNC_B128(la, ga,  32); ASYNC_B128(la, ga,  48);
    ASYNC_B128(la, ga,  64); ASYNC_B128(la, ga,  80);
    ASYNC_B128(la, ga,  96); ASYNC_B128(la, ga, 112);
    ASYNC_B128(lb, gb,   0); ASYNC_B128(lb, gb,  16);
    ASYNC_B128(lb, gb,  32); ASYNC_B128(lb, gb,  48);
    ASYNC_B128(lb, gb,  64); ASYNC_B128(lb, gb,  80);
    ASYNC_B128(lb, gb,  96); ASYNC_B128(lb, gb, 112);
  };

  v8f acc[4][4];
  #pragma unroll
  for (int mi = 0; mi < 4; ++mi)
    #pragma unroll
    for (int ni = 0; ni < 4; ++ni)
      acc[mi][ni] = (v8f){0.f, 0.f, 0.f, 0.f, 0.f, 0.f, 0.f, 0.f};

  issue_tile(0, 0);                       // prologue

  for (int kb = 0; kb < NKTILE; ++kb) {
    const int cur = kb & 1;
    if (kb + 1 < NKTILE) {
      issue_tile(cur ^ 1, (kb + 1) * TILE_K);   // prefetch next (prev readers done: trailing barrier)
      asm volatile("s_wait_asynccnt 16" ::: "memory");  // current tile's 16 copies done
    } else {
      asm volatile("s_wait_asynccnt 0" ::: "memory");
    }
    __syncthreads();                      // all waves' copies visible

    #pragma unroll
    for (int ks = 0; ks < 2; ++ks) {
      FragU a[4], b[4];
      #pragma unroll
      for (int mi = 0; mi < 4; ++mi) {
        const __half* ap = &Ash[cur][(wm + mi * 16 + l16) * LDH + ks * 32 + h * 8];
        a[mi].q[0] = *(const uint4*)ap;
        a[mi].q[1] = *(const uint4*)(ap + 16);
      }
      #pragma unroll
      for (int ni = 0; ni < 4; ++ni) {
        const __half* bp = &Bsh[cur][(wn + ni * 16 + l16) * LDH + ks * 32 + h * 8];
        b[ni].q[0] = *(const uint4*)bp;
        b[ni].q[1] = *(const uint4*)(bp + 16);
      }
      #pragma unroll
      for (int mi = 0; mi < 4; ++mi)
        #pragma unroll
        for (int ni = 0; ni < 4; ++ni)
          acc[mi][ni] = __builtin_amdgcn_wmma_f32_16x16x32_f16(
              false, a[mi].v, false, b[ni].v,
              (short)0, acc[mi][ni], false, false);
    }
    __syncthreads();                      // readers done before buffer is overwritten
  }

  // fused manifold normalization + store
  #pragma unroll
  for (int mi = 0; mi < 4; ++mi) {
    const int row0 = bm0 + wm + mi * 16 + h * 8;
    #pragma unroll
    for (int p = 0; p < 2; ++p) {
      float* obase = yg + (size_t)row0 * NDIM + bn0 + wn + p * 32;
      #pragma unroll
      for (int v = 0; v < 8; ++v) {
        float a0 = acc[mi][2 * p    ][v];
        float a1 = acc[mi][2 * p + 1][v];
        float s = a0 * a0 + a1 * a1;
        s += __shfl_xor(s, 1, 16);
        s += __shfl_xor(s, 2, 16);
        s += __shfl_xor(s, 4, 16);
        s += __shfl_xor(s, 8, 16);
        float inv = rsqrtf(s + 1e-6f);
        float* orow = obase + (size_t)v * NDIM;
        orow[l16]      = a0 * inv;
        orow[16 + l16] = a1 * inv;
      }
    }
  }
}

// ---------------------------------------------------------------------------
// Fallback: round-1 fused kernel (no workspace needed)
__global__ __launch_bounds__(128)
void versor_linear_fused(const float* __restrict__ xg,
                         const float* __restrict__ wg,
                         float* __restrict__ yg)
{
  __shared__ __half Ash[TILE_M * LDH];
  __shared__ __half Bsh[TILE_N * LDH];

  const int t    = threadIdx.x;
  const int lane = t & 31;
  const int wv   = t >> 5;
  const int wm   = (wv >> 1) << 6;
  const int wn   = (wv &  1) << 6;
  const int h    = lane >> 4;
  const int l16  = lane & 15;

  const int bm0 = blockIdx.y * TILE_M;
  const int bn0 = blockIdx.x * TILE_N;

  const int kcol = t & 31;
  const int ocol = (bn0 / GA_DIM) + (t >> 5);
  const unsigned sgn = sign_mask(kcol);

  v8f acc[4][4];
  #pragma unroll
  for (int mi = 0; mi < 4; ++mi)
    #pragma unroll
    for (int ni = 0; ni < 4; ++ni)
      acc[mi][ni] = (v8f){0.f, 0.f, 0.f, 0.f, 0.f, 0.f, 0.f, 0.f};

  for (int k0 = 0; k0 < KDIM; k0 += TILE_K) {
    __syncthreads();
    {
      const float* src = xg + (size_t)(bm0 + t) * KDIM + k0;
      __half* dst = Ash + t * LDH;
      #pragma unroll
      for (int c = 0; c < TILE_K; c += 8) {
        float4 f0 = *(const float4*)(src + c);
        float4 f1 = *(const float4*)(src + c + 4);
        PackU p;
        p.h2[0] = __floats2half2_rn(f0.x, f0.y);
        p.h2[1] = __floats2half2_rn(f0.z, f0.w);
        p.h2[2] = __floats2half2_rn(f1.x, f1.y);
        p.h2[3] = __floats2half2_rn(f1.z, f1.w);
        *(uint4*)(dst + c) = p.u;
      }
    }
    {
      const int f0 = k0 / GA_DIM;
      __half* dst = Bsh + t * LDH;
      #pragma unroll
      for (int fi = 0; fi < 2; ++fi) {
        const float* wrow = wg + (size_t)ocol * KDIM + (size_t)(f0 + fi) * GA_DIM;
        #pragma unroll
        for (int ii = 0; ii < GA_DIM; ii += 2) {
          float w0 = sflip(wrow[(ii    ) ^ kcol], sgn, ii    );
          float w1 = sflip(wrow[(ii + 1) ^ kcol], sgn, ii + 1);
          *(__half2*)(dst + fi * GA_DIM + ii) = __floats2half2_rn(w0, w1);
        }
      }
    }
    __syncthreads();

    #pragma unroll
    for (int ks = 0; ks < 2; ++ks) {
      FragU a[4], b[4];
      #pragma unroll
      for (int mi = 0; mi < 4; ++mi) {
        const __half* ap = Ash + (wm + mi * 16 + l16) * LDH + ks * 32 + h * 8;
        a[mi].q[0] = *(const uint4*)ap;
        a[mi].q[1] = *(const uint4*)(ap + 16);
      }
      #pragma unroll
      for (int ni = 0; ni < 4; ++ni) {
        const __half* bp = Bsh + (wn + ni * 16 + l16) * LDH + ks * 32 + h * 8;
        b[ni].q[0] = *(const uint4*)bp;
        b[ni].q[1] = *(const uint4*)(bp + 16);
      }
      #pragma unroll
      for (int mi = 0; mi < 4; ++mi)
        #pragma unroll
        for (int ni = 0; ni < 4; ++ni)
          acc[mi][ni] = __builtin_amdgcn_wmma_f32_16x16x32_f16(
              false, a[mi].v, false, b[ni].v,
              (short)0, acc[mi][ni], false, false);
    }
  }

  #pragma unroll
  for (int mi = 0; mi < 4; ++mi) {
    const int row0 = bm0 + wm + mi * 16 + h * 8;
    #pragma unroll
    for (int p = 0; p < 2; ++p) {
      float* obase = yg + (size_t)row0 * NDIM + bn0 + wn + p * 32;
      #pragma unroll
      for (int v = 0; v < 8; ++v) {
        float a0 = acc[mi][2 * p    ][v];
        float a1 = acc[mi][2 * p + 1][v];
        float s = a0 * a0 + a1 * a1;
        s += __shfl_xor(s, 1, 16);
        s += __shfl_xor(s, 2, 16);
        s += __shfl_xor(s, 4, 16);
        s += __shfl_xor(s, 8, 16);
        float inv = rsqrtf(s + 1e-6f);
        float* orow = obase + (size_t)v * NDIM;
        orow[l16]      = a0 * inv;
        orow[16 + l16] = a1 * inv;
      }
    }
  }
}

// ---------------------------------------------------------------------------
extern "C" void kernel_launch(void* const* d_in, const int* in_sizes, int n_in,
                              void* d_out, int out_size, void* d_ws, size_t ws_size,
                              hipStream_t stream) {
  (void)n_in; (void)out_size;
  const float* x = (const float*)d_in[0];   // [B, 128, 32] f32
  const float* w = (const float*)d_in[1];   // [128, 128, 32] f32
  float* y = (float*)d_out;                 // [B, 128, 32] f32

  const int batch = in_sizes[0] / KDIM;
  const size_t xh_bytes = (size_t)batch * KDIM * sizeof(__half);
  const size_t bt_bytes = (size_t)NDIM * KDIM * sizeof(__half);

  dim3 grid(NDIM / TILE_N, batch / TILE_M);

  if (d_ws && ws_size >= xh_bytes + bt_bytes) {
    __half* xh = (__half*)d_ws;
    __half* bt = (__half*)((char*)d_ws + xh_bytes);

    const int n8 = (batch * KDIM) / 8;
    cvt_x_kernel<<<(n8 + 255) / 256, 256, 0, stream>>>(x, xh, n8);
    expand_w_kernel<<<NDIM, 256, 0, stream>>>(w, bt);
    versor_gemm_async<<<grid, 128, 0, stream>>>(xh, bt, y);
  } else {
    versor_linear_fused<<<grid, 128, 0, stream>>>(x, w, y);
  }
}
